// FinerModel_40269613367308
// MI455X (gfx1250) — matrix operand order repaired and checked by reading, not compiled, not run on hardware
//
#include <hip/hip_runtime.h>
#include <hip/hip_bf16.h>

typedef _Float16 v16h __attribute__((ext_vector_type(16)));
typedef _Float16 v8h  __attribute__((ext_vector_type(8)));
typedef float    v8f  __attribute__((ext_vector_type(8)));

#define OMEGA 30.0f

__device__ __forceinline__ float finer_act(float z) {
    // FINER: sin(omega * (|z|+1) * z)
    float alpha = fabsf(z) + 1.0f;
    return __sinf(OMEGA * alpha * z);
}

// ---------------------------------------------------------------------------
// Kernel 1: convert hidden weights W1..W4 (f32 [256][256], row = out channel)
// into f16, pre-swizzled into the exact WMMA B-fragment (32x16 f16) lane
// layout so the main kernel's LDS reads are trivially addressed:
//   dst[((l*16+n)*8+k)*32 + lane][j]  (16 halves per lane, contiguous)
//   lane 0-15 : N=lane,    K = 32k + j (j<8), 32k + 8 + j (j>=8)
//   lane 16-31: N=lane-16, K = 32k + 8 + j (j<8), 32k + 16 + j (j>=8)
// ---------------------------------------------------------------------------
__global__ void finer_swizzle(const float* __restrict__ W1, const float* __restrict__ W2,
                              const float* __restrict__ W3, const float* __restrict__ W4,
                              _Float16* __restrict__ dst) {
    int d = blockIdx.x * blockDim.x + threadIdx.x;   // 0..262143
    int j    = d & 15;
    int lane = (d >> 4) & 31;
    int k    = (d >> 9) & 7;
    int n    = (d >> 12) & 15;
    int l    = d >> 16;
    const float* W = (l == 0) ? W1 : (l == 1) ? W2 : (l == 2) ? W3 : W4;
    int c  = 16 * n + (lane & 15);                   // output channel (B column)
    int kk = 32 * k + ((lane >> 4) << 3) + ((j < 8) ? j : (j + 8));   // input channel  (B row / K)
    dst[d] = (_Float16)W[c * 256 + kk];
}

// one async L2->LDS 16B transfer per lane; inst offset advances BOTH the
// global and LDS addresses (ISA 10.x async pseudocode), tracked by ASYNCcnt
#define ACOPY(OFF)                                                            \
    asm volatile("global_load_async_to_lds_b128 %0, %1, %2 offset:" #OFF     \
                 :: "v"(vdst), "v"(voff), "s"(gsrc) : "memory")

// ---------------------------------------------------------------------------
// Main kernel: 256 threads = 8 waves; each wave carries one 16-point tile
// through the whole network. Hidden-layer weights are staged one layer at a
// time into LDS (128 KB of the 320 KB WGP LDS) with async tensor-to-LDS DMA;
// activations live in registers as WMMA A-fragments, with an 8 KB/wave LDS
// scratch used only for the C-layout -> A-layout cross-lane transpose, which
// executes in the shadow of the async weight copy. The WMMA k-loop software-
// pipelines B-fragment ds_loads two deep so LDS latency is hidden.
// ---------------------------------------------------------------------------
__global__ void __launch_bounds__(256)
finer_main(const float* __restrict__ coords,
           const float* __restrict__ W0, const float* __restrict__ b0,
           const float* __restrict__ b1, const float* __restrict__ b2,
           const float* __restrict__ b3, const float* __restrict__ b4,
           const float* __restrict__ W5, const float* __restrict__ b5,
           const _Float16* __restrict__ Wswz,
           float* __restrict__ out, int nTiles, int N)
{
    extern __shared__ char smem[];
    _Float16* sW   = (_Float16*)smem;                        // 131072 B : one layer's B-fragments
    _Float16* sAct = (_Float16*)(smem + 131072);             //  65536 B : 8 waves x [16][256] f16
    float* sW0   = (float*)(smem + 131072 + 65536);          //   3072 B : W0 [256][3]
    float* sB0   = (float*)(smem + 131072 + 65536 + 3072);   //   1024 B
    float* sBias = (float*)(smem + 131072 + 65536 + 4096);   //   4096 B : b1..b4
    float* sW5   = (float*)(smem + 131072 + 65536 + 8192);   //   1024 B
    // total dynamic LDS = 205824 B

    const int tid  = threadIdx.x;
    const int wave = tid >> 5;
    const int lane = tid & 31;
    const int p    = lane & 15;   // point row within tile (A row), also B/C column index
    const int hi   = lane >> 4;
    const int koff = hi * 8;      // A-fragment K sub-offset for upper half-wave

    // stage small parameters
    for (int i = tid; i < 768; i += 256) sW0[i] = W0[i];
    if (tid < 256) {
        sB0[tid]         = b0[tid];
        sBias[tid]       = b1[tid];
        sBias[256 + tid] = b2[tid];
        sBias[512 + tid] = b3[tid];
        sBias[768 + tid] = b4[tid];
        sW5[tid]         = W5[tid];
    }

    const int  tile   = blockIdx.x * 8 + wave;
    const bool active = tile < nTiles;
    _Float16* sActW = sAct + wave * 4096;   // this wave's [16][256] scratch

    v16h a[8];                               // A-fragments: 16x256 activations, f16
    float x0 = 0.f, x1 = 0.f, x2 = 0.f;
    if (active) {
        int pt = tile * 16 + p;
        if (pt >= N) pt = N - 1;             // (tail safety; N is a multiple of 16 here)
        x0 = coords[3 * pt + 0];
        x1 = coords[3 * pt + 1];
        x2 = coords[3 * pt + 2];
    }

    // per-lane base of this lane's B-fragment slots (stride 512 halves per frag)
    const _Float16* wlane = sW + (size_t)lane * 16;

    for (int l = 0; l < 4; ++l) {
        __syncthreads();                     // previous layer done reading sW; staging visible
        // --- issue async copy of layer-l weights (pre-swizzled f16) into LDS ---
        {
            const char* gsrc = (const char*)(Wswz + (size_t)l * 65536);
            uint32_t voff = (uint32_t)tid * 16u;                       // per-lane global offset
            uint32_t vdst = (uint32_t)(uintptr_t)sW + voff;            // per-lane LDS byte addr
            ACOPY(0);      ACOPY(4096);   ACOPY(8192);   ACOPY(12288);
            ACOPY(16384);  ACOPY(20480);  ACOPY(24576);  ACOPY(28672);
            ACOPY(32768);  ACOPY(36864);  ACOPY(40960);  ACOPY(45056);
            ACOPY(49152);  ACOPY(53248);  ACOPY(57344);  ACOPY(61440);
            ACOPY(65536);  ACOPY(69632);  ACOPY(73728);  ACOPY(77824);
            ACOPY(81920);  ACOPY(86016);  ACOPY(90112);  ACOPY(94208);
            ACOPY(98304);  ACOPY(102400); ACOPY(106496); ACOPY(110592);
            ACOPY(114688); ACOPY(118784); ACOPY(122880); ACOPY(126976);
        }
        // --- A-fragment production overlaps the async DMA ---
        float bcs[16];
        if (active) {
            if (l == 0) {
                // layer 0 (3 -> 256): build A fragments directly in register layout.
                // A 16x32 f16 layout: lane holds row p; halves j=0..7 -> K=32k+koff+j,
                // j=8..15 -> K=32k+koff+16+(j-8).
                #pragma unroll
                for (int k = 0; k < 8; ++k) {
                    #pragma unroll
                    for (int j = 0; j < 16; ++j) {
                        int c = 32 * k + koff + ((j < 8) ? j : (j + 8));
                        float z = x0 * sW0[3 * c] + x1 * sW0[3 * c + 1] +
                                  x2 * sW0[3 * c + 2] + sB0[c];
                        a[k][j] = (_Float16)finer_act(z);
                    }
                }
            } else {
                // reload previous layer's activations as A-fragments (LDS transpose)
                #pragma unroll
                for (int k = 0; k < 8; ++k) {
                    const _Float16* base = sActW + p * 256 + 32 * k + koff;
                    v8h lo = *(const v8h*)base;
                    v8h hs = *(const v8h*)(base + 16);
                    a[k] = __builtin_shufflevector(lo, hs,
                        0, 1, 2, 3, 4, 5, 6, 7, 8, 9, 10, 11, 12, 13, 14, 15);
                }
            }
            // hoist all 16 bias values for this layer into registers so the
            // WMMA loop issues no scalar LDS loads (no dscnt==0 flushes)
            #pragma unroll
            for (int n = 0; n < 16; ++n) bcs[n] = sBias[l * 256 + 16 * n + p];
        }
        asm volatile("s_wait_asynccnt 0x0" ::: "memory");   // weight DMA complete
        __syncthreads();                                    // weights visible to all waves
        if (active) {
            // 2-deep software pipeline of B-fragment loads (frag f = n*8+k,
            // stride 512 halves); the +2 overrun reads land in sAct scratch
            // (inside our allocation) and are never consumed.
            v16h bcur = *(const v16h*)(wlane);
            v16h bnxt = *(const v16h*)(wlane + 512);
            #pragma unroll 2
            for (int n = 0; n < 16; ++n) {
                float bc = bcs[n];
                v8f acc = {bc, bc, bc, bc, bc, bc, bc, bc};
                #pragma unroll
                for (int k = 0; k < 8; ++k) {
                    v16h bfut = *(const v16h*)(wlane + (size_t)(n * 8 + k + 2) * 512);
                    acc = __builtin_amdgcn_wmma_f32_16x16x32_f16(
                        false, a[k], false, bcur, (short)0, acc, false, false);
                    bcur = bnxt;
                    bnxt = bfut;
                }
                // FINER activation on C tile, store f16 row-major [point][channel]
                #pragma unroll
                for (int r = 0; r < 8; ++r) {
                    float s = finer_act(acc[r]);
                    sActW[(r + hi * 8) * 256 + (16 * n + p)] = (_Float16)s;
                }
            }
        }
    }

    // final layer 256 -> 1: VALU dot product, cross-half shuffle reduce
    asm volatile("" ::: "memory");           // keep last layer's LDS stores before reads
    if (active) {
        const _Float16* row  = sActW + p * 256 + hi * 128;
        const float*    wrow = sW5 + hi * 128;
        float partial = 0.0f;
        #pragma unroll
        for (int j8 = 0; j8 < 16; ++j8) {
            v8h hv = *(const v8h*)(row + j8 * 8);
            #pragma unroll
            for (int e = 0; e < 8; ++e)
                partial += (float)hv[e] * wrow[j8 * 8 + e];
        }
        partial += __shfl_xor(partial, 16, 32);
        int pt = tile * 16 + p;
        if (hi == 0 && pt < N) out[pt] = partial + b5[0];
    }
}

// ---------------------------------------------------------------------------
extern "C" void kernel_launch(void* const* d_in, const int* in_sizes, int n_in,
                              void* d_out, int out_size, void* d_ws, size_t ws_size,
                              hipStream_t stream) {
    const float* coords = (const float*)d_in[0];
    const float* W0 = (const float*)d_in[1];
    const float* b0 = (const float*)d_in[2];
    const float* W1 = (const float*)d_in[3];
    const float* b1 = (const float*)d_in[4];
    const float* W2 = (const float*)d_in[5];
    const float* b2 = (const float*)d_in[6];
    const float* W3 = (const float*)d_in[7];
    const float* b3 = (const float*)d_in[8];
    const float* W4 = (const float*)d_in[9];
    const float* b4 = (const float*)d_in[10];
    const float* W5 = (const float*)d_in[11];
    const float* b5 = (const float*)d_in[12];
    float* out = (float*)d_out;
    _Float16* Wswz = (_Float16*)d_ws;        // 4 * 65536 f16 = 512 KB

    const int N       = in_sizes[0] / 3;     // 1,000,000 points
    const int nTiles  = (N + 15) / 16;       // 62,500
    const int nBlocks = (nTiles + 7) / 8;    // 7,813

    // pre-swizzle hidden-layer weights into B-fragment layout (f16)
    finer_swizzle<<<262144 / 256, 256, 0, stream>>>(W1, W2, W3, W4, Wswz);

    const size_t SMEM = 205824;              // 201 KB of the 320 KB WGP LDS
    (void)hipFuncSetAttribute(reinterpret_cast<const void*>(finer_main),
                              hipFuncAttributeMaxDynamicSharedMemorySize, (int)SMEM);
    finer_main<<<nBlocks, 256, SMEM, stream>>>(coords, W0, b0, b1, b2, b3, b4,
                                               W5, b5, Wswz, out, nTiles, N);
}
